// DeformSpatialAttLayer_80066780332223
// MI455X (gfx1250) — matrix operand order repaired and checked by reading, not compiled, not run on hardware
//
#include <hip/hip_runtime.h>

typedef __attribute__((ext_vector_type(16))) _Float16 v16h;
typedef __attribute__((ext_vector_type(8)))  _Float16 v8h;
typedef __attribute__((ext_vector_type(8)))  float    v8f;
typedef __attribute__((ext_vector_type(4)))  float    v4f;

#define BB   8
#define CC   256
#define HH   128
#define WW   128
#define HWSZ (HH*WW)          // 16384
#define KTAP 49               // 7x7 taps
#define NCH  147              // 3*49 conv output channels
#define NPAD 160              // 10 tiles of N=16
#define KDIM 98               // 2*49 im2col depth
#define KPAD 128              // 4 WMMA K-steps of 32

// ---------------- kernel 1: channel mean+max reduction -> y planes ----------
// Streams 134 MB of x exactly once: non-temporal loads keep L2 free for the
// small reused intermediates (y planes / packed B / gate).
__global__ void k_reduce(const float* __restrict__ x, float* __restrict__ yp) {
  int pix = blockIdx.x * blockDim.x + threadIdx.x;   // 0 .. B*HW-1
  int b  = pix >> 14;                                // / 16384
  int hw = pix & (HWSZ - 1);
  const float* xp = x + (size_t)b * CC * HWSZ + hw;
  float s = 0.f, m = -3.4e38f;
  #pragma unroll 4
  for (int c = 0; c < CC; ++c) {
    float v = __builtin_nontemporal_load(&xp[(size_t)c * HWSZ]);
    if (c + 16 < CC) __builtin_prefetch(&xp[(size_t)(c + 16) * HWSZ], 0, 0);
    s += v;
    m = fmaxf(m, v);
  }
  yp[((size_t)b * 2    ) * HWSZ + hw] = s * (1.0f / CC);
  yp[((size_t)b * 2 + 1) * HWSZ + hw] = m;
}

// ---------------- kernel 2: pack conv weights into WMMA B-layout (f16) ------
// B tile = 32(K) x 16(N). lanes 0-15 hold K=0..15 (N=lane), lanes 16-31 hold
// K=16..31 (N=lane-16); 16 halves per lane stored contiguously.
__global__ void k_packB(const float* __restrict__ w_off, _Float16* __restrict__ Bp) {
  int idx  = blockIdx.x * blockDim.x + threadIdx.x;  // 40*32*16 = 20480 total
  int j    = idx & 15;
  int lane = (idx >> 4) & 31;
  int st   = idx >> 9;                               // 0..39 = s*10 + t
  int t    = st % 10;
  int s    = st / 10;
  int K = s * 32 + ((lane & 16) ? 16 : 0) + j;
  int N = t * 16 + (lane & 15);
  float v = 0.f;
  if (K < KDIM && N < NCH) v = w_off[N * KDIM + K]; // w_off[n][c][ky][kx], K=c*49+ky*7+kx
  Bp[idx] = (_Float16)v;
}

// ---------------- kernel 3: fused WMMA conv + deform sampling + gate --------
__global__ void __launch_bounds__(128)
k_conv_sample(const float* __restrict__ yp, const _Float16* __restrict__ Bp,
              const float* __restrict__ b_off, const float* __restrict__ w_dcn,
              const float* __restrict__ b_dcn, const float* __restrict__ bn_g,
              const float* __restrict__ bn_b,  const float* __restrict__ bn_m,
              const float* __restrict__ bn_v,  float* __restrict__ gate) {
  __shared__ alignas(16) _Float16 Alds[4][16][KPAD]; // per-wave im2col, 16 KB
  __shared__ float off_lds[64][NPAD];                // conv outputs, 40 KB
  __shared__ float bias_lds[NPAD];
  __shared__ float wdcn_lds[2 * KTAP];
  __shared__ float part[128];

  const int tid  = threadIdx.x;
  const int wid  = tid >> 5;
  const int lane = tid & 31;

  // block -> (b, h, w0b); each block owns one row-segment of 64 pixels
  const int bw  = WW / 64;                            // 2
  int b   = blockIdx.x / (HH * bw);
  int rem = blockIdx.x % (HH * bw);
  int h   = rem / bw;
  int w0b = (rem % bw) * 64;
  int w0  = w0b + wid * 16;                           // this wave's 16 pixels

  for (int i = tid; i < NPAD; i += 128)     bias_lds[i] = (i < NCH) ? b_off[i] : 0.f;
  for (int i = tid; i < 2 * KTAP; i += 128) wdcn_lds[i] = w_dcn[i];

  // ---- im2col patch for 16 pixels into per-wave LDS (row-major 16 x 128) ---
  const float* ybase = yp + (size_t)b * 2 * HWSZ;
  for (int i = lane; i < 16 * KPAD; i += 32) {
    int p = i >> 7;
    int k = i & 127;
    float v = 0.f;
    if (k < KDIM) {
      int c  = k / KTAP;
      int r  = k % KTAP;
      int ky = r / 7, kx = r % 7;
      int yy = h + ky - 3;
      int xx = w0 + p + kx - 3;
      if (yy >= 0 && yy < HH && xx >= 0 && xx < WW)
        v = ybase[(size_t)c * HWSZ + yy * WW + xx];
    }
    Alds[wid][p][k] = (_Float16)v;
  }
  __syncthreads();

  // ---- load A fragments once (contiguous k = 16h + s, same as packed B), reuse across 10 N-tiles
  const int m = lane & 15;
  union { v16h v; v8h h[2]; } afr[4];
  #pragma unroll
  for (int s = 0; s < 4; ++s) {
    int klo = s * 32 + ((lane & 16) ? 16 : 0);       // lanes16-31: K 16-31
    afr[s].h[0] = *(const v8h*)&Alds[wid][m][klo];
    afr[s].h[1] = *(const v8h*)&Alds[wid][m][klo + 8];
  }

  // ---- 10 N-tiles x 4 K-steps of v_wmma_f32_16x16x32_f16 -------------------
  const int p0 = wid * 16;
  for (int t = 0; t < 10; ++t) {
    v8f acc = {};
    #pragma unroll
    for (int s = 0; s < 4; ++s) {
      v16h bfr = *(const v16h*)(Bp + (((size_t)(s * 10 + t) * 32 + lane) << 4));
      acc = __builtin_amdgcn_wmma_f32_16x16x32_f16(false, afr[s].v, false, bfr,
                                                   (short)0, acc, false, false);
    }
    // D layout: VGPR r -> (M=r, N=lane) lanes 0-15, (M=r+8, N=lane-16) lanes 16-31
    int n    = t * 16 + (lane & 15);
    int mofs = (lane & 16) ? 8 : 0;
    float bias = bias_lds[n];
    #pragma unroll
    for (int r = 0; r < 8; ++r)
      off_lds[p0 + r + mofs][n] = acc[r] + bias;
  }
  __syncthreads();

  // ---- deformable bilinear sampling + 49-tap einsum ------------------------
  // thread t handles pixel t%64, taps split half/half between thread pairs
  int p   = tid & 63;
  int khf = tid >> 6;
  int w   = w0b + p;
  const float* y0 = ybase;
  const float* y1 = ybase + HWSZ;
  float acc = 0.f;
  int kbeg = khf ? 25 : 0;
  int kend = khf ? 49 : 25;
  for (int k = kbeg; k < kend; ++k) {
    float dyv = off_lds[p][2 * k];
    float dxv = off_lds[p][2 * k + 1];
    float msk = 1.f / (1.f + __expf(-off_lds[p][98 + k]));
    float py = (float)h + (float)(k / 7 - 3) + dyv;
    float px = (float)w + (float)(k % 7 - 3) + dxv;
    float fy = floorf(py), fx = floorf(px);
    int iy0 = (int)fy, ix0 = (int)fx;
    float wy1 = py - fy, wx1 = px - fx;
    float wy0 = 1.f - wy1, wx0 = 1.f - wx1;
    int iy1 = iy0 + 1, ix1 = ix0 + 1;
    bool vy0 = (iy0 >= 0) && (iy0 < HH);
    bool vy1 = (iy1 >= 0) && (iy1 < HH);
    bool vx0 = (ix0 >= 0) && (ix0 < WW);
    bool vx1 = (ix1 >= 0) && (ix1 < WW);
    int o00 = iy0 * WW + ix0, o01 = iy0 * WW + ix1;
    int o10 = iy1 * WW + ix0, o11 = iy1 * WW + ix1;
    float v0 = 0.f, v1 = 0.f;
    if (vy0 && vx0) { float wgt = wy0 * wx0; v0 += wgt * y0[o00]; v1 += wgt * y1[o00]; }
    if (vy0 && vx1) { float wgt = wy0 * wx1; v0 += wgt * y0[o01]; v1 += wgt * y1[o01]; }
    if (vy1 && vx0) { float wgt = wy1 * wx0; v0 += wgt * y0[o10]; v1 += wgt * y1[o10]; }
    if (vy1 && vx1) { float wgt = wy1 * wx1; v0 += wgt * y0[o11]; v1 += wgt * y1[o11]; }
    acc += msk * (wdcn_lds[k] * v0 + wdcn_lds[KTAP + k] * v1);
  }
  part[tid] = acc;
  __syncthreads();

  if (tid < 64) {
    float s = part[tid] + part[tid + 64] + b_dcn[0];
    s = (s - bn_m[0]) * (bn_g[0] * rsqrtf(bn_v[0] + 1e-5f)) + bn_b[0];
    float g = 1.f / (1.f + __expf(-s));
    gate[(size_t)b * HWSZ + h * WW + w0b + tid] = g;
  }
}

// ---------------- kernel 4: broadcast gate over all channels ----------------
// 134 MB streamed out exactly once: non-temporal vector stores; gate reads
// stay temporal (reused 256x from L2).
__global__ void k_broadcast(const float* __restrict__ gate, v4f* __restrict__ out) {
  size_t i4 = (size_t)blockIdx.x * blockDim.x + threadIdx.x; // 8388608 total
  size_t e  = i4 << 2;
  int    b  = (int)(e >> 22);              // / (C*HW) = 2^22
  int    hw = (int)(e & (HWSZ - 1));       // hw is 4-aligned
  v4f g = *(const v4f*)&gate[(size_t)b * HWSZ + hw];
  __builtin_nontemporal_store(g, &out[i4]);
}

extern "C" void kernel_launch(void* const* d_in, const int* in_sizes, int n_in,
                              void* d_out, int out_size, void* d_ws, size_t ws_size,
                              hipStream_t stream) {
  const float* x     = (const float*)d_in[0];
  const float* w_off = (const float*)d_in[1];
  const float* b_off = (const float*)d_in[2];
  const float* w_dcn = (const float*)d_in[3];
  const float* b_dcn = (const float*)d_in[4];
  const float* bn_g  = (const float*)d_in[5];
  const float* bn_b  = (const float*)d_in[6];
  const float* bn_m  = (const float*)d_in[7];
  const float* bn_v  = (const float*)d_in[8];
  float* out = (float*)d_out;

  char* ws = (char*)d_ws;
  float*    yp   = (float*)ws;                           // B*2*HW f32   = 1 MB
  _Float16* Bp   = (_Float16*)(ws + (1 << 20));          // packed B     = 40 KB
  float*    gate = (float*)(ws + (1 << 20) + (1 << 16)); // B*HW f32     = 512 KB

  k_reduce     <<<(BB * HWSZ) / 256,     256, 0, stream>>>(x, yp);
  k_packB      <<<20480 / 256,           256, 0, stream>>>(w_off, Bp);
  k_conv_sample<<<BB * HH * (WW / 64),   128, 0, stream>>>(yp, Bp, b_off, w_dcn, b_dcn,
                                                           bn_g, bn_b, bn_m, bn_v, gate);
  k_broadcast  <<<(BB * CC * HWSZ / 4) / 256, 256, 0, stream>>>(gate, (v4f*)out);
}
